// ScaledDotProductAttention_50972671869111
// MI455X (gfx1250) — matrix operand order repaired and checked by reading, not compiled, not run on hardware
//
#include <hip/hip_runtime.h>

typedef __attribute__((ext_vector_type(16))) _Float16 v16h;
typedef __attribute__((ext_vector_type(8)))  float    v8f;
typedef __attribute__((ext_vector_type(4)))  float    v4f;

#define Bz 8
#define Hz 16
#define Lz 1024
#define DKz 64
#define DVz 64
#define SROW 1028   // padded score-row stride in floats (bank-conflict-free WMMA C store)

// Dynamic LDS layout:
//   s_scores : 16 x SROW f32     (65792 B)
//   s_vt     : 32 x 64   f16     ( 4096 B)
//   s_rinv   : 16        f32     (   64 B)
#define SMEM_BYTES (16*SROW*4 + 32*64*2 + 16*4)

extern "C" __global__ __launch_bounds__(32)
void attn_wmma_kernel(const float* __restrict__ Q, const float* __restrict__ K,
                      const float* __restrict__ V, const float* __restrict__ PK,
                      const float* __restrict__ PV,
                      float* __restrict__ ctx_out, float* __restrict__ attn_out)
{
    extern __shared__ char smem[];
    float*    s_scores = (float*)smem;
    _Float16* s_vt     = (_Float16*)(smem + 16*SROW*4);
    float*    s_rinv   = (float*)(smem + 16*SROW*4 + 32*64*2);

    const int lane = threadIdx.x;     // 0..31 (wave32)
    const int hi   = lane >> 4;       // half-wave id
    const int lr   = lane & 15;

    const int wg = blockIdx.x;
    const int qt = wg & 63;           // L/16 = 64 q-tiles per head
    const int bh = wg >> 6;
    const int h  = bh & (Hz - 1);
    const int q0 = qt << 4;

    const float* Qb  = Q  + ((size_t)bh * Lz) * DKz;
    const float* Kb  = K  + ((size_t)bh * Lz) * DKz;
    const float* Vb  = V  + ((size_t)bh * Lz) * DVz;
    const float* PKb = PK + ((size_t)h  * Lz) * DKz;
    const float* PVb = PV + ((size_t)h  * Lz) * DVz;

    // ---- Q tile as two A-fragments (dk 0..31, 32..63); 1/sqrt(64) folded in ----
    // 16-bit A 16x32 layout: element e -> k = 2*(e&8) + (e&7) + 8*hi, row M = lr
    const float* qrow = Qb + (size_t)(q0 + lr) * DKz;
    v16h a0, a1;
    #pragma unroll
    for (int e = 0; e < 16; ++e) {
        int dk = ((e & 8) << 1) + (e & 7) + (hi << 3);
        a0[e] = (_Float16)(qrow[dk]      * 0.125f);
        a1[e] = (_Float16)(qrow[dk + 32] * 0.125f);
    }

    // ---- Phase 1: scores = (Q/8) * (K+PK)^T  ->  LDS (16 x 1024 f32) ----
    for (int kt = 0; kt < Lz / 16; ++kt) {
        // 16-bit B 32x16 layout: element e -> k = 2*(e&8) + (e&7) + 8*hi, col N = lr
        const int krow = kt * 16 + lr;
        const float* kp = Kb  + (size_t)krow * DKz;
        const float* pk = PKb + (size_t)krow * DKz;
        v16h b0, b1;
        #pragma unroll
        for (int e = 0; e < 16; ++e) {
            int dk = ((e & 8) << 1) + (e & 7) + (hi << 3);
            b0[e] = (_Float16)(kp[dk]      + pk[dk]);
            b1[e] = (_Float16)(kp[dk + 32] + pk[dk + 32]);
        }
        v8f c = {};
        c = __builtin_amdgcn_wmma_f32_16x16x32_f16(false, a0, false, b0, (short)0, c, false, false);
        c = __builtin_amdgcn_wmma_f32_16x16x32_f16(false, a1, false, b1, (short)0, c, false, false);
        // C layout: lane<16 -> (M=r, N=lr); lane>=16 -> (M=r+8, N=lr)
        #pragma unroll
        for (int r = 0; r < 8; ++r)
            s_scores[(r + (hi << 3)) * SROW + kt * 16 + lr] = c[r];
    }
    __syncthreads();

    // ---- Phase 2: row softmax; coalesced b128 attn writes; keep exp() in LDS ----
    float* attn_row0 = attn_out + (size_t)bh * Lz * Lz + (size_t)q0 * Lz;
    for (int r = 0; r < 16; ++r) {
        float* srow = &s_scores[r * SROW];
        float m = -3.0e38f;
        #pragma unroll
        for (int i = 0; i < 8; ++i) {
            v4f x = *(const v4f*)&srow[i * 128 + lane * 4];
            m = fmaxf(m, fmaxf(fmaxf(x.x, x.y), fmaxf(x.z, x.w)));
        }
        #pragma unroll
        for (int off = 16; off > 0; off >>= 1)
            m = fmaxf(m, __shfl_xor(m, off, 32));
        float s = 0.f;
        #pragma unroll
        for (int i = 0; i < 8; ++i) {
            float* p = &srow[i * 128 + lane * 4];
            v4f x = *(v4f*)p;
            x.x = __expf(x.x - m); x.y = __expf(x.y - m);
            x.z = __expf(x.z - m); x.w = __expf(x.w - m);
            *(v4f*)p = x;
            s += (x.x + x.y) + (x.z + x.w);
        }
        #pragma unroll
        for (int off = 16; off > 0; off >>= 1)
            s += __shfl_xor(s, off, 32);
        float inv = 1.0f / s;
        if (lane == 0) s_rinv[r] = inv;
        float* arow = attn_row0 + (size_t)r * Lz;
        #pragma unroll
        for (int i = 0; i < 8; ++i) {
            v4f x = *(const v4f*)&srow[i * 128 + lane * 4];
            x *= inv;
            *(v4f*)&arow[i * 128 + lane * 4] = x;   // coalesced 512B/instr
        }
    }
    __syncthreads();

    // ---- Phase 3: context = (exp_scores @ (V+PV)) * rinv ----
    v8f acc[4] = {};
    for (int kc = 0; kc < Lz / 32; ++kc) {
        // Stage (V+PV)[kc*32 .. +32) x 64 into LDS as f16 (coalesced reads)
        #pragma unroll
        for (int j = 0; j < 16; ++j) {
            int row = j * 2 + hi;
            int col = lr * 4;
            int k = kc * 32 + row;
            v4f x = *(const v4f*)(Vb  + (size_t)k * DVz + col);
            v4f y = *(const v4f*)(PVb + (size_t)k * DVz + col);
            x += y;
            _Float16* d = &s_vt[row * 64 + col];
            d[0] = (_Float16)x.x; d[1] = (_Float16)x.y;
            d[2] = (_Float16)x.z; d[3] = (_Float16)x.w;
        }
        __syncthreads();
        // A fragment: unnormalized exp scores -> f16 on the fly
        v16h af;
        #pragma unroll
        for (int e = 0; e < 16; ++e) {
            int kl = ((e & 8) << 1) + (e & 7) + (hi << 3);
            af[e] = (_Float16)s_scores[lr * SROW + kc * 32 + kl];
        }
        #pragma unroll
        for (int nt = 0; nt < 4; ++nt) {
            v16h bf;
            #pragma unroll
            for (int e = 0; e < 16; ++e) {
                int kl = ((e & 8) << 1) + (e & 7) + (hi << 3);
                bf[e] = s_vt[kl * 64 + nt * 16 + lr];
            }
            acc[nt] = __builtin_amdgcn_wmma_f32_16x16x32_f16(false, af, false, bf,
                                                             (short)0, acc[nt], false, false);
        }
        __syncthreads();
    }

    // ---- Write context, folding in 1/rowsum ----
    #pragma unroll
    for (int r = 0; r < 8; ++r) {
        int row = r + (hi << 3);
        float inv = s_rinv[row];
        float* crow = ctx_out + ((size_t)bh * Lz + q0 + row) * DVz;
        #pragma unroll
        for (int nt = 0; nt < 4; ++nt)
            crow[nt * 16 + lr] = acc[nt][r] * inv;
    }
}

extern "C" void kernel_launch(void* const* d_in, const int* in_sizes, int n_in,
                              void* d_out, int out_size, void* d_ws, size_t ws_size,
                              hipStream_t stream) {
    const float* Q  = (const float*)d_in[0];
    const float* K  = (const float*)d_in[1];
    const float* V  = (const float*)d_in[2];
    const float* PK = (const float*)d_in[3];
    const float* PV = (const float*)d_in[4];
    float* ctx  = (float*)d_out;
    float* attn = (float*)d_out + (size_t)Bz * Hz * Lz * DVz;  // tuple order: (context, attn)

    dim3 grid(Bz * Hz * (Lz / 16));  // 8192 single-wave workgroups
    attn_wmma_kernel<<<grid, dim3(32), SMEM_BYTES, stream>>>(Q, K, V, PK, PV, ctx, attn);
}
